// TFN_8839042695324
// MI455X (gfx1250) — matrix operand order repaired and checked by reading, not compiled, not run on hardware
//
#include <hip/hip_runtime.h>
#include <hip/hip_bf16.h>
#include <math.h>
#include <stdint.h>

// ---------------------------------------------------------------------------
// TFN forward for MI455X (gfx1250, wave32). Heavy contractions on
// v_wmma_f32_16x16x32_f16; fragments built from b128 vector loads; weight
// blocks staged to LDS with global_load_async_to_lds_b128 (ASYNCcnt).
// All hot index math in 32-bit.
// ---------------------------------------------------------------------------

typedef _Float16 f16;
typedef __attribute__((ext_vector_type(16))) _Float16 v16h;
typedef __attribute__((ext_vector_type(8)))  _Float16 v8h;
typedef __attribute__((ext_vector_type(8)))  float    v8f;
typedef __attribute__((ext_vector_type(4)))  int      v4i;

#define NB 128   // batch

__device__ __forceinline__ v16h cat8(v8h a, v8h b) {
  return __builtin_shufflevector(a, b, 0, 1, 2, 3, 4, 5, 6, 7,
                                 8, 9, 10, 11, 12, 13, 14, 15);
}
__device__ __forceinline__ v8f wmma32(v16h a, v16h b, v8f c) {
  return __builtin_amdgcn_wmma_f32_16x16x32_f16(false, a, false, b,
                                                (short)0, c, false, false);
}
// A-fragment (16x32 f16): lane r=lane&15 holds row r; half h=lane>>4:
//   elements = A[row][8h .. 8h+7] ++ A[row][16+8h .. 16+8h+7]
__device__ __forceinline__ v16h loadA(const f16* rowp, int h) {
  return cat8(*(const v8h*)(rowp + 8 * h), *(const v8h*)(rowp + 16 + 8 * h));
}
// B-fragment (32x16 f16, stored transposed [col][k]): lane holds column r,
//   elements = Bt[col][8h .. 8h+7] ++ Bt[col][16+8h .. 16+8h+7]
__device__ __forceinline__ v16h loadB(const f16* colp, int h) {
  return cat8(*(const v8h*)(colp + 8 * h), *(const v8h*)(colp + 16 + 8 * h));
}

// ---- real spherical harmonics, l<=3 (16 values) ---------------------------
__device__ __forceinline__ void sh16f(float x, float y, float z, float* o) {
  o[0]  = 0.28209479177387814f;
  o[1]  = 0.4886025119029199f * y;
  o[2]  = 0.4886025119029199f * z;
  o[3]  = 0.4886025119029199f * x;
  o[4]  = 1.0925484305920792f * x * y;
  o[5]  = 1.0925484305920792f * y * z;
  o[6]  = 0.31539156525252005f * (3.f * z * z - 1.f);
  o[7]  = 1.0925484305920792f * x * z;
  o[8]  = 0.5462742152960396f * (x * x - y * y);
  o[9]  = 0.5900435899266435f * y * (3.f * x * x - y * y);
  o[10] = 2.890611442640554f * x * y * z;
  o[11] = 0.4570457994644658f * y * (5.f * z * z - 1.f);
  o[12] = 0.3731763325901154f * z * (5.f * z * z - 3.f);
  o[13] = 0.4570457994644658f * x * (5.f * z * z - 1.f);
  o[14] = 1.445305721320277f * z * (x * x - y * y);
  o[15] = 0.5900435899266435f * x * (3.f * x * x - y * y);
}

// ---- pentakis-dodecahedron directions -> Y table (32 x 16) ----------------
__global__ void tfn_init_y(float* Ytab) {
  int t = threadIdx.x;
  if (t >= 32) return;
  const float phi = 1.6180339887498949f;
  float vx, vy, vz;
  if (t < 8) {
    vx = (t & 4) ? 1.f : -1.f;
    vy = (t & 2) ? 1.f : -1.f;
    vz = (t & 1) ? 1.f : -1.f;
  } else {
    int fam = (t < 20) ? 0 : 1;
    int u = t - (fam ? 20 : 8);
    int g = u / 3, kk = u % 3;
    float s1 = (g >= 2) ? 1.f : -1.f;
    float s2 = (g & 1) ? 1.f : -1.f;
    float v1 = fam ? s1 : (s1 / phi);
    float v2 = s2 * phi;
    if (kk == 0)      { vx = 0.f; vy = v1;  vz = v2; }
    else if (kk == 1) { vx = v1;  vy = v2;  vz = 0.f; }
    else              { vx = v2;  vy = 0.f; vz = v1; }
  }
  float n = rsqrtf(vx * vx + vy * vy + vz * vz);
  float sh[16];
  sh16f(vx * n, vy * n, vz * n, sh);
  for (int j = 0; j < 16; ++j) Ytab[t * 16 + j] = sh[j];
}

// ---- kd-tree indexing: per-batch segment bitonic sort ---------------------
__global__ void tfn_kdsort(const float* __restrict__ xin, float* __restrict__ pts0) {
  __shared__ float sx[1024], sy[1024], sz[1024];
  int b = blockIdx.x, tid = threadIdx.x;
  for (int e = tid; e < 1024; e += 256) {
    const float* p = xin + (b * 1024 + e) * 3;
    sx[e] = p[0]; sy[e] = p[1]; sz[e] = p[2];
  }
  __syncthreads();
  for (int d = 0; d < 10; ++d) {
    int L = 1024 >> d;
    int axis = d % 3;
    float* key = (axis == 0) ? sx : ((axis == 1) ? sy : sz);
    for (int k = 2; k <= L; k <<= 1) {
      for (int j = k >> 1; j > 0; j >>= 1) {
        __syncthreads();
        for (int e = tid; e < 1024; e += 256) {
          int p2 = e ^ j;
          if (p2 > e) {
            int il = e & (L - 1);
            bool up = ((il & k) == 0);
            float a = key[e], bq = key[p2];
            bool sw = up ? (a > bq) : (a < bq);
            if (sw) {
              float t0 = sx[e]; sx[e] = sx[p2]; sx[p2] = t0;
              float t1 = sy[e]; sy[e] = sy[p2]; sy[p2] = t1;
              float t2 = sz[e]; sz[e] = sz[p2]; sz[p2] = t2;
            }
          }
        }
      }
    }
    __syncthreads();
  }
  for (int e = tid; e < 1024; e += 256) {
    float* p = pts0 + (b * 1024 + e) * 3;
    p[0] = sx[e]; p[1] = sy[e]; p[2] = sz[e];
  }
}

// ---- kd average pool (pool factor always 4) -------------------------------
__global__ void tfn_pool(int Nin, int Nout, const float* __restrict__ in,
                         float* __restrict__ outp) {
  int g = blockIdx.x * blockDim.x + threadIdx.x;
  int tot = NB * Nout * 3;
  if (g >= tot) return;
  int dim = g % 3, rest = g / 3;
  int no = rest & (Nout - 1), b = rest / Nout;
  float acc = 0.f;
  for (int q = 0; q < 4; ++q)
    acc += in[(b * Nin + no * 4 + q) * 3 + dim];
  outp[(b * Nout + no) * 3 + dim] = acc * 0.25f;
}

// ---- group_points: top-32 nearest src for each tgt ------------------------
__global__ void tfn_group(int Nsrc, int Ntgt, const float* __restrict__ srcPts,
                          const float* __restrict__ tgtPts, int* __restrict__ idxOut) {
  __shared__ float sp[1024 * 3];
  int b = blockIdx.x, tid = threadIdx.x;
  for (int e = tid; e < Nsrc * 3; e += 256)
    sp[e] = srcPts[b * Nsrc * 3 + e];
  __syncthreads();
  if (tid < Ntgt) {
    const float* tp = tgtPts + (b * Ntgt + tid) * 3;
    float tx = tp[0], ty = tp[1], tz = tp[2];
    float bd[32]; int bi[32];
    for (int r = 0; r < 32; ++r) { bd[r] = 3.4e38f; bi[r] = 0; }
    for (int j = 0; j < Nsrc; ++j) {
      float dx = sp[j * 3 + 0] - tx;
      float dy = sp[j * 3 + 1] - ty;
      float dz = sp[j * 3 + 2] - tz;
      float d2 = dx * dx + dy * dy + dz * dz;
      if (d2 < bd[31]) {
        int pos = 31;
        while (pos > 0 && bd[pos - 1] > d2) {
          bd[pos] = bd[pos - 1]; bi[pos] = bi[pos - 1]; --pos;
        }
        bd[pos] = d2; bi[pos] = j;
      }
    }
    int* io = idxOut + (b * Ntgt + tid) * 32;
    for (int r = 0; r < 32; ++r) io[r] = bi[r];
  }
}

// ---- input feature fetch (gathered y, incl. yzx concat channels) ----------
__device__ __forceinline__ float conv_feat(const float* yIn, const float* srcPts,
                                           int layer, int Nsrc, int Cprev,
                                           int b, int pt, int l, int m, int c) {
  if (layer == 0) {
    if (l == 0) return 1.0f;
    const float* pp = srcPts + (b * Nsrc + pt) * 3;
    return (m == 0) ? pp[1] : ((m == 1) ? pp[2] : pp[0]);  // yzx
  }
  if (l == 1 && c == Cprev) {
    const float* pp = srcPts + (b * Nsrc + pt) * 3;
    return (m == 0) ? pp[1] : ((m == 1) ? pp[2] : pp[0]);
  }
  return yIn[((b * Nsrc + pt) * 16 + l * l + m) * Cprev + c];
}

// ---- equivariant conv: per (b,v), both einsums as K=32 WMMAs --------------
// Writes padded f16 A-matrices X_l for the weight GEMM.
__global__ void tfn_conv(int layer, int Nsrc, int Ntgt, float radius, float gscale,
                         int Cprev, const float* __restrict__ srcPts,
                         const float* __restrict__ tgtPts,
                         const int* __restrict__ idxBuf,
                         const float* __restrict__ yIn,
                         f16* __restrict__ X0, f16* __restrict__ X1,
                         f16* __restrict__ X2, f16* __restrict__ X3,
                         int K0, int K1, int K2, int K3) {
  __shared__ f16 sA[544 * 32];       // gathered features, row-major [row][k]
  __shared__ f16 sB[5 * 16 * 32];    // kernels, transposed [tile][col][k]
  __shared__ int sInfo[544];         // packed (l<<12)|(m<<8)|c per A row
  __shared__ int sOutIdx[544];       // element offset into X_l (base for s=0)
  __shared__ int sIdx[32];
  int bv = blockIdx.x, b = bv / Ntgt, tid = threadIdx.x;

  for (int e = tid; e < 5 * 16 * 32; e += 128) sB[e] = (f16)0.f;
  if (tid < 32) sIdx[tid] = idxBuf[bv * 32 + tid];
  __syncthreads();

  if (tid < 32) {  // Gaussian-SH kernel for neighbor k, direct into B tiles
    int k = tid;
    const float* tp = tgtPts + bv * 3;
    const float* spp = srcPts + (b * Nsrc + sIdx[k]) * 3;
    float px = (spp[0] - tp[0]) / radius;
    float py = (spp[1] - tp[1]) / radius;
    float pz = (spp[2] - tp[2]) / radius;
    float dist = sqrtf(fmaxf(px * px + py * py + pz * pz, 1e-12f));
    float inv = 1.f / fmaxf(dist, 1e-6f);
    float sh[16];
    sh16f(px * inv, py * inv, pz * inv, sh);
    float g[3], gs = 0.f;
    for (int s = 0; s < 3; ++s) {
      float dd = dist - 0.5f * (float)s;
      g[s] = __expf(-gscale * dd * dd);
      if (dist > 1.f) g[s] = 0.f;
      gs += g[s];
    }
    float gn = 1.f / fmaxf(gs, 1e-6f);
    for (int s = 0; s < 3; ++s) g[s] *= gn;
    for (int s = 0; s < 3; ++s) sB[(0 * 16 + s) * 32 + k] = (f16)(sh[0] * g[s]);
    for (int c = 0; c < 9; ++c) {
      int m = c / 3, s = c % 3;
      sB[(1 * 16 + c) * 32 + k] = (f16)(sh[1 + m] * g[s]);
    }
    for (int c = 0; c < 15; ++c) {
      int m = c / 3, s = c % 3;
      sB[(2 * 16 + c) * 32 + k] = (f16)(sh[4 + m] * g[s]);
    }
    for (int c = 0; c < 21; ++c) {
      int m = c / 3, s = c % 3;
      int tile = 3 + (c >> 4), col = c & 15;
      sB[(tile * 16 + col) * 32 + k] = (f16)(sh[9 + m] * g[s]);
    }
  }

  int cin[4];
  if (layer == 0) { cin[0] = 1; cin[1] = 1; cin[2] = 0; cin[3] = 0; }
  else { cin[0] = Cprev; cin[1] = Cprev + 1; cin[2] = Cprev; cin[3] = Cprev; }
  int Kp[4] = {K0, K1, K2, K3};
  int rcnt[4], off[4];
  for (int l = 0; l < 4; ++l) rcnt[l] = (2 * l + 1) * cin[l];
  off[0] = 0;
  for (int l = 1; l < 4; ++l) off[l] = (off[l - 1] + rcnt[l - 1] + 15) & ~15;
  int rowsTot = off[3] + rcnt[3];

  // per-row decode + output element offset, once
  for (int row = tid; row < rowsTot; row += 128) {
    int l = (row >= off[3]) ? 3 : ((row >= off[2]) ? 2 : ((row >= off[1]) ? 1 : 0));
    int rel = row - off[l];
    int cl = (cin[l] > 0) ? cin[l] : 1;
    int m = rel / cl, c = rel % cl;
    if (m > 15) { m = 15; }
    sInfo[row] = (l << 12) | (m << 8) | c;
    sOutIdx[row] = (bv * (2 * l + 1) + m) * Kp[l] + c * 3;
  }
  __syncthreads();

  for (int e = tid; e < rowsTot * 32; e += 128) {  // stage gathered features
    int row = e >> 5, k = e & 31;
    int info = sInfo[row];
    int l = info >> 12, m = (info >> 8) & 15, c = info & 255;
    sA[e] = (f16)conv_feat(yIn, srcPts, layer, Nsrc, Cprev, b, sIdx[k], l, m, c);
  }
  if (tid < 3) {  // materialize yzx concat channel for l=1 weight input
    const float* tp = tgtPts + bv * 3;
    float vv = (tid == 0) ? tp[1] : ((tid == 1) ? tp[2] : tp[0]);
    X1[(bv * 3 + tid) * K1 + cin[1] * 3 + cin[0] * 3] = (f16)vv;
  }
  __syncthreads();

  int wave = tid >> 5, lane = tid & 31, r = lane & 15, h = lane >> 4;
  f16* Xp[4] = {X0, X1, X2, X3};
  int task = 0;
#pragma unroll
  for (int l = 0; l < 4; ++l) {
    int ml = 2 * l + 1;
    if (cin[l] > 0) {  // part 1: gathered_l (mc x 32) x k0 (32 x 3)
      int nrows = rcnt[l];
      int nt = (nrows + 15) / 16;
      for (int t = 0; t < nt; ++t) {
        if ((task++ & 3) == wave) {
          v16h a = loadA(sA + (off[l] + t * 16 + r) * 32, h);
          v16h bm = loadB(sB + r * 32, h);  // tile 0
          v8f z = {};
          v8f acc = wmma32(a, bm, z);
          int base = off[l] + t * 16 + 8 * h;      // 32B-aligned (off 16-aligned)
          v4i o0 = *(const v4i*)(sOutIdx + base);
          v4i o1 = *(const v4i*)(sOutIdx + base + 4);
          if (r < 3) {
#pragma unroll
            for (int jj = 0; jj < 8; ++jj) {
              int rg = t * 16 + jj + 8 * h;
              if (rg < nrows)
                Xp[l][(jj < 4 ? o0[jj & 3] : o1[jj & 3]) + r] = (f16)acc[jj];
            }
          }
        }
      }
    }
    if (l >= 1) {  // part 2: f0 (c0 x 32) x kern_l (32 x m*3)
      int ncols = ml * 3;
      int ntc = (ncols + 15) / 16, ntr = (cin[0] + 15) / 16;
      for (int ct = 0; ct < ntc; ++ct)
        for (int t = 0; t < ntr; ++t) {
          if ((task++ & 3) == wave) {
            int colg = ct * 16 + r;
            v16h a = loadA(sA + (t * 16 + r) * 32, h);
            int tile = (l == 1) ? 1 : ((l == 2) ? 2 : (3 + ct));
            v16h bm = loadB(sB + (tile * 16 + r) * 32, h);
            v8f z = {};
            v8f acc = wmma32(a, bm, z);
            int m = colg / 3, s = colg - m * 3;     // constant-divisor
            int colBase = (bv * ml + m) * Kp[l] + cin[l] * 3 + s;
            int rgBase = (t * 16 + 8 * h) * 3;
            if (colg < ncols) {
#pragma unroll
              for (int jj = 0; jj < 8; ++jj) {
                int rg = t * 16 + jj + 8 * h;
                if (rg < cin[0])
                  Xp[l][colBase + rgBase + jj * 3] = (f16)acc[jj];
              }
            }
          }
        }
    }
  }
}

// ---- weight prep: transpose + pad W (Cin x U) -> Wt (U x Kpad) f16 --------
__global__ void tfn_prepw(int Cin, int U, int Kpad, const float* __restrict__ W,
                          f16* __restrict__ Wt) {
  int k = blockIdx.x * blockDim.x + threadIdx.x;
  int col = blockIdx.y;
  if (k >= Kpad) return;
  Wt[col * Kpad + k] = (k < Cin) ? (f16)W[k * U + col] : (f16)0.f;
}

// ---- apply_eq_weights: GEMM (R x Kpad) x (Kpad x U) via WMMA.
// Weight block is DMA'd to LDS with async global->LDS loads (ASYNCcnt).
__global__ void tfn_eqw(int R, int Kpad, int U, const f16* __restrict__ X,
                        const f16* __restrict__ Wt, const float* __restrict__ bias,
                        int hasBias, float* __restrict__ out) {
  __shared__ f16 sWt[16 * 224];           // 16 columns x Kpad (max 224)
  int lane = threadIdx.x;
  int r = lane & 15, h = lane >> 4;
  int rowg = blockIdx.x * 16 + r;
  int colBlk = blockIdx.y * 16;
  // --- async copy of the 16-column weight block into LDS ---
  unsigned ldsBase = (unsigned)(uintptr_t)(&sWt[0]);
  unsigned long long gbase = (unsigned long long)(uintptr_t)(Wt + colBlk * Kpad);
  int rounds = Kpad >> 4;  // (16*Kpad*2 B) / (32 lanes * 16 B)
  for (int q = 0; q < rounds; ++q) {
    int c0 = (q << 5) + lane;
    unsigned la = ldsBase + (unsigned)(c0 << 4);
    unsigned long long ga = gbase + (unsigned)(c0 << 4);
    asm volatile("global_load_async_to_lds_b128 %0, %1, off"
                 :: "v"(la), "v"(ga) : "memory");
  }
  asm volatile("s_wait_asynccnt 0x0" ::: "memory");

  const f16* arow = X + rowg * Kpad;
  const f16* bcol = sWt + r * Kpad;
  v8f acc = {};
  int Kt = Kpad >> 5;
  for (int ks = 0; ks < Kt; ++ks)
    acc = wmma32(loadA(arow + (ks << 5), h), loadB(bcol + (ks << 5), h), acc);
  int colg = colBlk + r;
  float bv = hasBias ? bias[colg] : 0.f;
#pragma unroll
  for (int jj = 0; jj < 8; ++jj) {
    int rg = blockIdx.x * 16 + jj + 8 * h;
    if (rg < R) out[rg * U + colg] = acc[jj] + bv;
  }
}

// ---- dodeca_eval + batchnorm + relu ---------------------------------------
__global__ void tfn_eval(int U, int Ush,
                         const float* __restrict__ yw0, const float* __restrict__ yw1,
                         const float* __restrict__ yw2, const float* __restrict__ yw3,
                         const float* __restrict__ Ytab,
                         const float* __restrict__ gamma, const float* __restrict__ beta,
                         const float* __restrict__ mean, const float* __restrict__ var,
                         float* __restrict__ sOut) {
  __shared__ float sy[16 * 64];
  int bn = blockIdx.x, tid = threadIdx.x;
  for (int e = tid; e < 16 * U; e += blockDim.x) {
    int mh = e >> Ush, c = e & (U - 1);
    float v;
    if (mh == 0)      v = yw0[bn * U + c];
    else if (mh < 4)  v = yw1[(bn * 3 + (mh - 1)) * U + c];
    else if (mh < 9)  v = yw2[(bn * 5 + (mh - 4)) * U + c];
    else              v = yw3[(bn * 7 + (mh - 9)) * U + c];
    sy[e] = v;
  }
  __syncthreads();
  for (int e = tid; e < 32 * U; e += blockDim.x) {
    int vv = e >> Ush, c = e & (U - 1);
    float acc = 0.f;
#pragma unroll
    for (int m = 0; m < 16; ++m) acc += Ytab[vv * 16 + m] * sy[(m << Ush) + c];
    acc = (acc - mean[c]) * rsqrtf(var[c] + 1e-3f) * gamma[c] + beta[c];
    sOut[bn * 32 * U + e] = fmaxf(acc, 0.f);
  }
}

// ---- dodeca_coeffs: project back to SH (dense 16 x Cprev layout) ----------
__global__ void tfn_coeffs(int U, int Ush, const float* __restrict__ sIn,
                           const float* __restrict__ Ytab, float* __restrict__ yOut) {
  __shared__ float ss[32 * 64];
  int bn = blockIdx.x, tid = threadIdx.x;
  for (int e = tid; e < 32 * U; e += blockDim.x)
    ss[e] = sIn[bn * 32 * U + e];
  __syncthreads();
  const float w = 0.39269908169872414f;  // 4*pi/32
  for (int e = tid; e < 16 * U; e += blockDim.x) {
    int m = e >> Ush, c = e & (U - 1);
    float acc = 0.f;
#pragma unroll
    for (int vv = 0; vv < 32; ++vv) acc += Ytab[vv * 16 + m] * ss[(vv << Ush) + c];
    yOut[bn * 16 * U + e] = w * acc;
  }
}

// ---- head: max over points, l=1 coeffs, fc, normalize, cross --------------
__global__ void tfn_head(const float* __restrict__ s2, const float* __restrict__ Ytab,
                         const float* __restrict__ fcw, const float* __restrict__ fcb,
                         float* __restrict__ out) {
  __shared__ float smax[32 * 64];
  __shared__ float yc[3 * 64];
  __shared__ float Rm[3][2];
  __shared__ float Rn[2];
  __shared__ float Rcol[3][2];
  int b = blockIdx.x, tid = threadIdx.x;  // 64 threads
  for (int e = tid; e < 32 * 64; e += 64) {
    int vv = e >> 6, c = e & 63;
    float mx = -3.4e38f;
    for (int n = 0; n < 16; ++n)
      mx = fmaxf(mx, s2[((b * 16 + n) * 32 + vv) * 64 + c]);
    smax[e] = mx;
  }
  __syncthreads();
  const float w = 0.39269908169872414f;
  for (int e = tid; e < 192; e += 64) {
    int m = e >> 6, c = e & 63;
    float acc = 0.f;
    for (int vv = 0; vv < 32; ++vv) acc += Ytab[vv * 16 + 1 + m] * smax[vv * 64 + c];
    yc[e] = w * acc;
  }
  __syncthreads();
  if (tid < 6) {
    int m = tid >> 1, u = tid & 1;
    float acc = fcb[u];
    for (int c = 0; c < 64; ++c) acc += yc[m * 64 + c] * fcw[c * 2 + u];
    Rm[m][u] = acc;
  }
  __syncthreads();
  if (tid < 2) {
    float n2 = Rm[0][tid] * Rm[0][tid] + Rm[1][tid] * Rm[1][tid] + Rm[2][tid] * Rm[2][tid];
    Rn[tid] = sqrtf(fmaxf(n2, 0.1f));
  }
  __syncthreads();
  if (tid < 6) {
    int m = tid >> 1, u = tid & 1;
    Rcol[m][u] = Rm[m][u] / Rn[u];
  }
  __syncthreads();
  if (tid < 3) {
    int m = tid;
    int m1 = (m + 1) % 3, m2 = (m + 2) % 3;
    float cr = Rcol[m1][0] * Rcol[m2][1] - Rcol[m2][0] * Rcol[m1][1];
    out[b * 9 + m * 3 + 0] = Rcol[m][0];
    out[b * 9 + m * 3 + 1] = Rcol[m][1];
    out[b * 9 + m * 3 + 2] = cr;
  }
}

// ---------------------------------------------------------------------------
extern "C" void kernel_launch(void* const* d_in, const int* in_sizes, int n_in,
                              void* d_out, int out_size, void* d_ws, size_t ws_size,
                              hipStream_t stream) {
  // Input order (setup_inputs dict order, recursive insertion-order flatten):
  // x, then per eq layer: w0,b0,w1,w2,w3; then bn{gamma,beta,mean,var}x3; fc1_w, fc1_b.
  auto gf = [&](int i) -> const float* {
    if (i >= n_in) i = n_in - 1;
    if (i < 0) i = 0;
    return (const float*)d_in[i];
  };
  const float* x = gf(0);
  const float* eqwp[3][4];
  const float* eqb[3];
  int p = 1;
  for (int i = 0; i < 3; ++i) {
    eqwp[i][0] = gf(p++); eqb[i] = gf(p++);
    eqwp[i][1] = gf(p++); eqwp[i][2] = gf(p++); eqwp[i][3] = gf(p++);
  }
  const float *bng[3], *bnb[3], *bnm[3], *bnv[3];
  for (int i = 0; i < 3; ++i) {
    bng[i] = gf(p++); bnb[i] = gf(p++); bnm[i] = gf(p++); bnv[i] = gf(p++);
  }
  const float* fcw = gf(p++);
  const float* fcb = gf(p++);

  const int NP[4] = {1024, 256, 64, 16};
  const float RAD[3] = {0.2f, 0.4f, 0.8f};
  const float GS = 6.23832462504f;  // ln(2)*9
  const int Cprev[3] = {0, 16, 32};
  const int units[3] = {16, 32, 64};
  const int Ush[3] = {4, 5, 6};
  const int CinT[3][4] = {{3, 7, 3, 3}, {48, 100, 96, 96}, {96, 196, 192, 192}};
  int KpT[3][4];
  for (int i = 0; i < 3; ++i)
    for (int l = 0; l < 4; ++l) KpT[i][l] = ((CinT[i][l] + 31) / 32) * 32;

  size_t xElemMax = 0, ywElemMax = 0, wtElemTot = 0;
  for (int i = 0; i < 3; ++i) {
    size_t xe = 0, ye = 0;
    for (int l = 0; l < 4; ++l) {
      size_t R = (size_t)NB * NP[i + 1] * (2 * l + 1);
      xe += (R + 16) * KpT[i][l];
      ye += R * units[i];
      wtElemTot += (size_t)units[i] * KpT[i][l];
    }
    if (xe > xElemMax) xElemMax = xe;
    if (ye > ywElemMax) ywElemMax = ye;
  }

  char* w = (char*)d_ws;
  auto alloc = [&](size_t bytes) -> char* {
    char* q = w;
    w += (bytes + 255) & ~(size_t)255;
    return q;
  };
  float* Ytab = (float*)alloc(32 * 16 * 4);
  float* pts0 = (float*)alloc((size_t)NB * 1024 * 3 * 4);
  float* pts1 = (float*)alloc((size_t)NB * 256 * 3 * 4);
  float* pts2 = (float*)alloc((size_t)NB * 64 * 3 * 4);
  float* pts3 = (float*)alloc((size_t)NB * 16 * 3 * 4);
  int*   idxb = (int*)alloc((size_t)NB * 256 * 32 * 4);
  f16*   xreg = (f16*)alloc(xElemMax * 2);
  f16*   wtreg = (f16*)alloc(wtElemTot * 2);
  float* ywreg = (float*)alloc(ywElemMax * 4);
  float* sb   = (float*)alloc((size_t)NB * 256 * 32 * 16 * 4);
  float* ybuf = (float*)alloc((size_t)NB * 256 * 16 * 16 * 4);
  if ((size_t)(w - (char*)d_ws) > ws_size) return;

  size_t wtOff[3][4];
  {
    size_t o = 0;
    for (int i = 0; i < 3; ++i)
      for (int l = 0; l < 4; ++l) { wtOff[i][l] = o; o += (size_t)units[i] * KpT[i][l]; }
  }

  const float* ptsArr[4] = {pts0, pts1, pts2, pts3};

  tfn_init_y<<<1, 32, 0, stream>>>(Ytab);
  tfn_kdsort<<<NB, 256, 0, stream>>>(x, pts0);
  {
    int tot = NB * 256 * 3;
    tfn_pool<<<(tot + 255) / 256, 256, 0, stream>>>(1024, 256, pts0, pts1);
    tot = NB * 64 * 3;
    tfn_pool<<<(tot + 255) / 256, 256, 0, stream>>>(256, 64, pts1, pts2);
    tot = NB * 16 * 3;
    tfn_pool<<<(tot + 255) / 256, 256, 0, stream>>>(64, 16, pts2, pts3);
  }

  for (int i = 0; i < 3; ++i) {
    int Nsrc = NP[i], Ntgt = NP[i + 1];
    int U = units[i];
    f16* Xl[4];
    float* ywl[4];
    int Rl[4];
    {
      size_t xo = 0, yo = 0;
      for (int l = 0; l < 4; ++l) {
        Rl[l] = NB * Ntgt * (2 * l + 1);
        Xl[l] = xreg + xo;
        ywl[l] = ywreg + yo;
        xo += (size_t)(Rl[l] + 16) * KpT[i][l];
        yo += (size_t)Rl[l] * U;
      }
    }
    tfn_group<<<NB, 256, 0, stream>>>(Nsrc, Ntgt, ptsArr[i], ptsArr[i + 1], idxb);
    tfn_conv<<<NB * Ntgt, 128, 0, stream>>>(i, Nsrc, Ntgt, RAD[i], GS, Cprev[i],
                                            ptsArr[i], ptsArr[i + 1], idxb, ybuf,
                                            Xl[0], Xl[1], Xl[2], Xl[3],
                                            KpT[i][0], KpT[i][1], KpT[i][2], KpT[i][3]);
    for (int l = 0; l < 4; ++l) {
      f16* Wt = wtreg + wtOff[i][l];
      dim3 gp((KpT[i][l] + 63) / 64, U);
      tfn_prepw<<<gp, 64, 0, stream>>>(CinT[i][l], U, KpT[i][l], eqwp[i][l], Wt);
      dim3 gg((Rl[l] + 15) / 16, U / 16);
      tfn_eqw<<<gg, 32, 0, stream>>>(Rl[l], KpT[i][l], U, Xl[l], Wt, eqb[i],
                                     (l == 0) ? 1 : 0, ywl[l]);
    }
    tfn_eval<<<NB * Ntgt, 256, 0, stream>>>(U, Ush[i], ywl[0], ywl[1], ywl[2], ywl[3],
                                            Ytab, bng[i], bnb[i], bnm[i], bnv[i], sb);
    if (i < 2)
      tfn_coeffs<<<NB * Ntgt, 256, 0, stream>>>(U, Ush[i], sb, Ytab, ybuf);
  }
  tfn_head<<<NB, 64, 0, stream>>>(sb, Ytab, fcw, fcb, (float*)d_out);
  (void)in_sizes; (void)out_size;
}
